// DecomposedCandidateBaseline_32779190403194
// MI455X (gfx1250) — matrix operand order repaired and checked by reading, not compiled, not run on hardware
//
#include <hip/hip_runtime.h>
#include <hip/hip_bf16.h>
#include <math.h>

// Problem constants (match reference)
#define Bn     2048
#define LCc    20
#define LXx    20
#define LHh    200
#define Dd     128
#define Hh     512
#define DENSEe 256
#define QIN    (3*Dd + Hh)   // 896
#define FIN    (7*Dd + Hh)   // 1408

// gfx1250 is wave32; lanes split into halves of 16 per the WMMA layouts.
typedef __attribute__((ext_vector_type(16))) __bf16 v16bf;
typedef __attribute__((ext_vector_type(8)))  float  v8f;
typedef __attribute__((ext_vector_type(4)))  unsigned int v4u;
typedef __attribute__((ext_vector_type(4)))  int v4i;
typedef __attribute__((ext_vector_type(8)))  int v8i;
typedef __attribute__((ext_vector_type(2)))  __bf16 v2bf;

#ifdef __has_builtin
#if __has_builtin(__builtin_amdgcn_cvt_pk_bf16_f32)
#define HAVE_PK_BF16 1
#endif
#if __has_builtin(__builtin_amdgcn_tensor_load_to_lds)
#define HAVE_TDM 1
#endif
#endif

union AF { v16bf v; unsigned u[8]; };

__device__ __forceinline__ float silu_f(float x) { return x / (1.0f + __expf(-x)); }

// pack two floats into two bf16.
//  - hardware pack op if the builtin exists (not on current toolchain)
//  - else a single v_perm_b32 grabbing the two high halves (bf16 truncation,
//    standard inference down-convert; 1 VALU op instead of ~6)
__device__ __forceinline__ unsigned pk2(float a, float b) {
#ifdef HAVE_PK_BF16
  v2bf p = __builtin_amdgcn_cvt_pk_bf16_f32(a, b);
  return __builtin_bit_cast(unsigned, p);
#else
  // result bytes [3:2] = b.bytes[3:2], [1:0] = a.bytes[3:2]
  return __builtin_amdgcn_perm(__float_as_uint(b), __float_as_uint(a), 0x07060302u);
#endif
}
// float -> bf16 scalar (truncation)
__device__ __forceinline__ unsigned short f2bf(float f) {
  return (unsigned short)(__float_as_uint(f) >> 16);
}
__device__ __forceinline__ v8f wmma_bf16(const AF& a, const AF& b, v8f c) {
  // D = A(16x32 bf16) * B(32x16 bf16) + C(16x16 f32)
  return __builtin_amdgcn_wmma_f32_16x16x32_bf16(false, a.v, false, b.v,
                                                 (short)0, c, false, false);
}

// -----------------------------------------------------------------------------
// Kernel 1: per-history-row pipeline. One workgroup (8 waves) per batch row.
//   time_h = silu(t*w1+b1) @ w2 + b2                 (WMMA, K=128)
//   hist_emb = tok+pos+grp+time_h                     (fused into epilogue)
//   comp   = silu(LN([grp|time_h] @ comp_w + cb))     (WMMA, K=256)
//   comp_hist = masked mean over L                    (LDS atomics)
// -----------------------------------------------------------------------------
__global__ __launch_bounds__(256) void k_history(
    const int* __restrict__ h_tok, const int* __restrict__ h_pos,
    const int* __restrict__ h_grp, const int* __restrict__ h_msk,
    const float* __restrict__ h_time,
    const float* __restrict__ tok_emb, const float* __restrict__ pos_emb,
    const float* __restrict__ grp_emb,
    const float* __restrict__ tw1, const float* __restrict__ tb1,
    const float* __restrict__ tw2, const float* __restrict__ tb2,
    const float* __restrict__ cw,  const float* __restrict__ cb,
    const float* __restrict__ cg,  const float* __restrict__ cbt,
    float* __restrict__ hist_emb, float* __restrict__ comp_hist)
{
  const int b   = blockIdx.x;
  const int tid = threadIdx.x;
  const int wv  = tid >> 5, ln = tid & 31;
  const int lm  = ln & 15,  hf = ln >> 4;

  __shared__ unsigned       sW2[64][128];        // time_w2 bf16 pairs [k/2][n]   32KB
  __shared__ unsigned       sCW[128][128];       // comp_w  bf16 pairs [k/2][n]   64KB
  __shared__ unsigned short sTH[8][16][128];     // time_h bf16, per-wave tile    32KB
  __shared__ float sGrp[4][128];
  __shared__ float sW1[128], sB1[128], sB2[128];
  __shared__ float sCb[128], sCg[128], sCbt[128];
  __shared__ float sCsum[128];
  __shared__ float sCnt;

  for (int i = tid; i < 64*128; i += 256) {
    int k2 = i >> 7, n = i & 127;
    sW2[k2][n] = pk2(tw2[(2*k2)*Dd + n], tw2[(2*k2+1)*Dd + n]);
  }
  for (int i = tid; i < 128*128; i += 256) {
    int k2 = i >> 7, n = i & 127;
    sCW[k2][n] = pk2(cw[(2*k2)*Dd + n], cw[(2*k2+1)*Dd + n]);
  }
  for (int i = tid; i < 4*128; i += 256) sGrp[i >> 7][i & 127] = grp_emb[i];
  if (tid < 128) {
    sW1[tid]=tw1[tid]; sB1[tid]=tb1[tid]; sB2[tid]=tb2[tid];
    sCb[tid]=cb[tid];  sCg[tid]=cg[tid];  sCbt[tid]=cbt[tid];
    sCsum[tid]=0.f;
  }
  if (tid == 0) sCnt = 0.f;
  __syncthreads();
  if (tid < LHh) atomicAdd(&sCnt, (float)h_msk[b*LHh + tid]);

  float csum[8];
  #pragma unroll
  for (int nt = 0; nt < 8; nt++) csum[nt] = 0.f;

  for (int r = 0; r < 2; r++) {               // 13 M-tiles over 8 waves
    const int  tile = r*8 + wv;
    const int  l    = tile*16 + lm;
    const bool rv   = (tile < 13) && (l < LHh);
    const float tv  = rv ? h_time[b*LHh + l]        : 0.f;
    const int   gv  = rv ? h_grp [b*LHh + l]        : 0;
    const float mv  = rv ? (float)h_msk[b*LHh + l]  : 0.f;
    const int   tkv = rv ? h_tok [b*LHh + l]        : 0;
    const int   psv = rv ? h_pos [b*LHh + l]        : 0;

    if (tile < 13) {                          // wave-uniform predicate
      // ---------------- time GEMM: [16x128] @ [128x128] ----------------
      v8f accT[8] = {};
      #pragma unroll
      for (int kt = 0; kt < 4; kt++) {
        AF A;
        #pragma unroll
        for (int v = 0; v < 8; v++) {
          int k0 = kt*32 + 2*v + (hf ? 16 : 0);
          A.u[v] = pk2(silu_f(tv*sW1[k0]   + sB1[k0]),
                       silu_f(tv*sW1[k0+1] + sB1[k0+1]));
        }
        #pragma unroll
        for (int nt = 0; nt < 8; nt++) {
          AF Bf;
          #pragma unroll
          for (int v = 0; v < 8; v++) Bf.u[v] = sW2[kt*16 + hf*8 + v][nt*16 + lm];
          accT[nt] = wmma_bf16(A, Bf, accT[nt]);
        }
      }
      // epilogue: stage time_h (bf16) + fused history-embedding gather/store
      #pragma unroll
      for (int nt = 0; nt < 8; nt++) {
        #pragma unroll
        for (int rr = 0; rr < 8; rr++) {
          int m = rr + hf*8;
          int n = nt*16 + lm;
          float th = accT[nt][rr] + sB2[n];
          sTH[wv][m][n] = f2bf(th);
          int mt = __shfl(tkv, m, 32);
          int mp = __shfl(psv, m, 32);
          int mg = __shfl(gv,  m, 32);
          int gl = tile*16 + m;
          if (gl < LHh) {
            float e = tok_emb[(size_t)mt*Dd + n] + pos_emb[(size_t)mp*Dd + n]
                    + sGrp[mg][n] + th;
            hist_emb[((size_t)b*LHh + gl)*Dd + n] = e;
          }
        }
      }
      // wave-local LDS RAW on sTH (DS ops are in-order per wave)
      asm volatile("s_wait_dscnt 0" ::: "memory");

      // ---------------- comp GEMM: [16x256] @ [256x128] ----------------
      v8f accC[8] = {};
      #pragma unroll
      for (int kt = 0; kt < 8; kt++) {
        AF A;
        if (kt < 4) {                         // K 0..127 : grp_h
          #pragma unroll
          for (int v = 0; v < 8; v++) {
            int k0 = kt*32 + 2*v + (hf ? 16 : 0);
            A.u[v] = pk2(sGrp[gv][k0], sGrp[gv][k0+1]);
          }
        } else {                              // K 128..255 : time_h
          #pragma unroll
          for (int v = 0; v < 8; v++) {
            int k0 = (kt-4)*32 + 2*v + (hf ? 16 : 0);
            A.u[v] = (unsigned)sTH[wv][lm][k0] | ((unsigned)sTH[wv][lm][k0+1] << 16);
          }
        }
        #pragma unroll
        for (int nt = 0; nt < 8; nt++) {
          AF Bf;
          #pragma unroll
          for (int v = 0; v < 8; v++) Bf.u[v] = sCW[kt*16 + hf*8 + v][nt*16 + lm];
          accC[nt] = wmma_bf16(A, Bf, accC[nt]);
        }
      }
      // epilogue: LayerNorm (cross-lane within half) + SiLU + masked sum
      #pragma unroll
      for (int rr = 0; rr < 8; rr++) {
        float vals[8]; float s1 = 0.f, s2 = 0.f;
        #pragma unroll
        for (int nt = 0; nt < 8; nt++) {
          float x = accC[nt][rr] + sCb[nt*16 + lm];
          vals[nt] = x; s1 += x; s2 += x*x;
        }
        #pragma unroll
        for (int o = 1; o < 16; o <<= 1) {
          s1 += __shfl_xor(s1, o, 32);
          s2 += __shfl_xor(s2, o, 32);
        }
        float mean = s1 * (1.f/128.f);
        float var  = s2 * (1.f/128.f) - mean*mean;
        float rstd = rsqrtf(var + 1e-5f);
        int   m    = rr + hf*8;
        float mm   = __shfl(mv, m, 32);
        #pragma unroll
        for (int nt = 0; nt < 8; nt++) {
          int n = nt*16 + lm;
          float y = (vals[nt] - mean)*rstd*sCg[n] + sCbt[n];
          csum[nt] += mm * silu_f(y);
        }
      }
    }
  }
  #pragma unroll
  for (int nt = 0; nt < 8; nt++) atomicAdd(&sCsum[nt*16 + lm], csum[nt]);
  __syncthreads();
  if (tid < 128) comp_hist[(size_t)b*Dd + tid] = sCsum[tid] / fmaxf(sCnt, 1.f);
}

// -----------------------------------------------------------------------------
// Kernel 2: candidate / context masked-mean gathers (bandwidth bound).
// -----------------------------------------------------------------------------
__global__ __launch_bounds__(128) void k_summaries(
    const int* __restrict__ cand_t, const int* __restrict__ cand_m,
    const int* __restrict__ ctx_t,  const int* __restrict__ ctx_m,
    const float* __restrict__ tok_emb,
    float* __restrict__ cand_s, float* __restrict__ ctx_s)
{
  int b = blockIdx.x, d = threadIdx.x;
  float s = 0.f, c = 0.f;
  for (int l = 0; l < LCc; l++)
    if (cand_m[b*LCc + l]) { s += tok_emb[(size_t)cand_t[b*LCc + l]*Dd + d]; c += 1.f; }
  cand_s[(size_t)b*Dd + d] = s / fmaxf(c, 1.f);
  s = 0.f; c = 0.f;
  for (int l = 0; l < LXx; l++)
    if (ctx_m[b*LXx + l]) { s += tok_emb[(size_t)ctx_t[b*LXx + l]*Dd + d]; c += 1.f; }
  ctx_s[(size_t)b*Dd + d] = s / fmaxf(c, 1.f);
}

// -----------------------------------------------------------------------------
// Kernel 3: dense_s = silu(LN(dense @ dense_w + b)).  16-row tiles, VALU.
// -----------------------------------------------------------------------------
__global__ __launch_bounds__(256) void k_dense(
    const float* __restrict__ x, const float* __restrict__ w,
    const float* __restrict__ bias, const float* __restrict__ g,
    const float* __restrict__ bt, float* __restrict__ dense_s)
{
  int b0 = blockIdx.x*16, tid = threadIdx.x;
  __shared__ float sx[16][256];
  __shared__ float rs1[16], rs2[16];
  for (int i = tid; i < 16*256; i += 256)
    sx[i >> 8][i & 255] = x[(size_t)(b0 + (i >> 8))*DENSEe + (i & 255)];
  if (tid < 16) { rs1[tid] = 0.f; rs2[tid] = 0.f; }
  __syncthreads();
  float a0[16], a1[16];
  #pragma unroll
  for (int r2 = 0; r2 < 16; r2++) { a0[r2] = 0.f; a1[r2] = 0.f; }
  for (int k = 0; k < DENSEe; k++) {
    float w0 = w[(size_t)k*Hh + tid];
    float w1 = w[(size_t)k*Hh + tid + 256];
    #pragma unroll
    for (int r2 = 0; r2 < 16; r2++) { a0[r2] += sx[r2][k]*w0; a1[r2] += sx[r2][k]*w1; }
  }
  float bb0 = bias[tid], bb1 = bias[tid + 256];
  #pragma unroll
  for (int r2 = 0; r2 < 16; r2++) {
    a0[r2] += bb0; a1[r2] += bb1;
    atomicAdd(&rs1[r2], a0[r2] + a1[r2]);
    atomicAdd(&rs2[r2], a0[r2]*a0[r2] + a1[r2]*a1[r2]);
  }
  __syncthreads();
  float g0 = g[tid], g1 = g[tid + 256], t0 = bt[tid], t1 = bt[tid + 256];
  #pragma unroll
  for (int r2 = 0; r2 < 16; r2++) {
    float mean = rs1[r2]*(1.f/512.f);
    float var  = rs2[r2]*(1.f/512.f) - mean*mean;
    float rstd = rsqrtf(var + 1e-5f);
    dense_s[(size_t)(b0 + r2)*Hh + tid]       = silu_f((a0[r2]-mean)*rstd*g0 + t0);
    dense_s[(size_t)(b0 + r2)*Hh + tid + 256] = silu_f((a1[r2]-mean)*rstd*g1 + t1);
  }
}

// -----------------------------------------------------------------------------
// Kernel 4: query = silu(LN(q_in @ query_w + b)).  16-row tiles, VALU.
// -----------------------------------------------------------------------------
__global__ __launch_bounds__(128) void k_query(
    const float* __restrict__ cand_s, const float* __restrict__ ctx_s,
    const float* __restrict__ comp_hist, const float* __restrict__ dense_s,
    const float* __restrict__ qw, const float* __restrict__ qb,
    const float* __restrict__ qg, const float* __restrict__ qbt,
    float* __restrict__ query)
{
  int b0 = blockIdx.x*16, tid = threadIdx.x;
  __shared__ float sq[16][QIN];
  __shared__ float rs1[16], rs2[16];
  for (int i = tid; i < 16*QIN; i += 128) {
    int r = i / QIN, c = i % QIN;
    float v;
    if      (c < 128) v = cand_s   [(size_t)(b0 + r)*Dd + c];
    else if (c < 256) v = ctx_s    [(size_t)(b0 + r)*Dd + (c - 128)];
    else if (c < 384) v = comp_hist[(size_t)(b0 + r)*Dd + (c - 256)];
    else              v = dense_s  [(size_t)(b0 + r)*Hh + (c - 384)];
    sq[r][c] = v;
  }
  if (tid < 16) { rs1[tid] = 0.f; rs2[tid] = 0.f; }
  __syncthreads();
  float a[16];
  #pragma unroll
  for (int r2 = 0; r2 < 16; r2++) a[r2] = 0.f;
  for (int k = 0; k < QIN; k++) {
    float wv = qw[(size_t)k*Dd + tid];
    #pragma unroll
    for (int r2 = 0; r2 < 16; r2++) a[r2] += sq[r2][k]*wv;
  }
  float bb = qb[tid];
  #pragma unroll
  for (int r2 = 0; r2 < 16; r2++) {
    a[r2] += bb;
    atomicAdd(&rs1[r2], a[r2]);
    atomicAdd(&rs2[r2], a[r2]*a[r2]);
  }
  __syncthreads();
  float gg = qg[tid], tt = qbt[tid];
  #pragma unroll
  for (int r2 = 0; r2 < 16; r2++) {
    float mean = rs1[r2]*(1.f/128.f);
    float var  = rs2[r2]*(1.f/128.f) - mean*mean;
    query[(size_t)(b0 + r2)*Dd + tid] = silu_f((a[r2]-mean)*rsqrtf(var+1e-5f)*gg + tt);
  }
}

// -----------------------------------------------------------------------------
// Kernel 5: masked attention pooling + fused-feature assembly.
// hist_emb tile (200x128 f32, 100KB) is staged into LDS ONCE per block (it is
// read twice: scores + weighted sum). When available, the stage is done by the
// Tensor Data Mover (tensor_load_to_lds + s_wait_tensorcnt); otherwise by
// cooperative vector loads.
// -----------------------------------------------------------------------------
__global__ __launch_bounds__(256) void k_attention(
    const float* __restrict__ hist_emb, const float* __restrict__ query,
    const int* __restrict__ h_msk,
    const float* __restrict__ cand_s, const float* __restrict__ ctx_s,
    const float* __restrict__ comp_hist, const float* __restrict__ dense_s,
    float* __restrict__ fused)
{
  int b = blockIdx.x, tid = threadIdx.x;
  const int wv = tid >> 5, ln = tid & 31;
  __shared__ float sH[LHh][Dd];      // 100KB history-embedding tile
  __shared__ float sQ[Dd];
  __shared__ float sS[LHh];
  __shared__ float sSum;

  const float* src = hist_emb + (size_t)b*LHh*Dd;
#ifdef HAVE_TDM
  if (tid < 32) {   // one wave issues the DMA (TDM is a wave-level op)
    unsigned lds_lo = (unsigned)(uintptr_t)&sH[0][0];
    unsigned long long ga = (unsigned long long)(uintptr_t)src;
    // D# group 0: count=1, lds_addr, global_addr[56:0], type=2 ("image")
    v4u g0;
    g0[0] = 1u;
    g0[1] = lds_lo;
    g0[2] = (unsigned)(ga & 0xFFFFFFFFull);
    g0[3] = (unsigned)((ga >> 32) & 0x01FFFFFFull) | (2u << 30);
    // D# group 1: data_size=4B, 2D tensor 128x200, tile 128x200, stride 128
    v8i g1;
    g1[0] = (int)(2u << 16);                       // wg_mask=0, data_size=2 (4B)
    g1[1] = (int)((unsigned)(Dd  & 0xFFFF) << 16); // tensor_dim0[15:0]=128
    g1[2] = (int)((unsigned)(LHh & 0xFFFF) << 16); // tensor_dim1[15:0]=200
    g1[3] = (int)((unsigned)(Dd  & 0xFFFF) << 16); // tile_dim0=128
    g1[4] = (int)(LHh & 0xFFFF);                   // tile_dim1=200, tile_dim2=0
    g1[5] = (int)Dd;                               // tensor_dim0_stride=128
    g1[6] = 0; g1[7] = 0;                          // tensor_dim1_stride unused
    v4i gz = {0, 0, 0, 0};
#if __clang_major__ >= 23
    v8i gz8 = {0, 0, 0, 0, 0, 0, 0, 0};
    __builtin_amdgcn_tensor_load_to_lds(g0, g1, gz, gz, gz8, 0);
#else
    __builtin_amdgcn_tensor_load_to_lds(g0, g1, gz, gz, 0);
#endif
    __builtin_amdgcn_s_wait_tensorcnt(0);
  }
#else
  for (int i = tid; i < LHh*Dd; i += 256) sH[i >> 7][i & 127] = src[i];
#endif
  if (tid < Dd) sQ[tid] = query[(size_t)b*Dd + tid];
  __syncthreads();

  // scores: wave-cooperative dot products (conflict-free LDS access)
  for (int l = wv; l < LHh; l += 8) {
    float s = 0.f;
    #pragma unroll
    for (int j = 0; j < 4; j++) s += sH[l][ln + 32*j]*sQ[ln + 32*j];
    #pragma unroll
    for (int o = 16; o >= 1; o >>= 1) s += __shfl_xor(s, o, 32);
    if (ln == 0)
      sS[l] = (h_msk[b*LHh + l] > 0) ? s*0.08838834764831845f : -1e9f;
  }
  __syncthreads();
  if (tid == 0) {
    float mx = -3.0e38f;
    for (int l = 0; l < LHh; l++) mx = fmaxf(mx, sS[l]);
    float sm = 0.f;
    for (int l = 0; l < LHh; l++) { float e = __expf(sS[l]-mx); sS[l] = e; sm += e; }
    sSum = sm;
  }
  __syncthreads();
  if (tid < Dd) {
    int d = tid;
    float h = 0.f;
    for (int l = 0; l < LHh; l++) h += sS[l]*sH[l][d];
    h /= sSum;
    float cs = cand_s[(size_t)b*Dd + d];
    float xs = ctx_s [(size_t)b*Dd + d];
    float ch = comp_hist[(size_t)b*Dd + d];
    size_t fo = (size_t)b*FIN;
    fused[fo + d]       = cs;
    fused[fo + 128 + d] = xs;
    fused[fo + 256 + d] = h;
    fused[fo + 384 + d] = ch;
    fused[fo + 512 + d] = cs*h;
    fused[fo + 640 + d] = cs*ch;
    fused[fo + 768 + d] = fabsf(h - ch);
  }
  for (int k = tid; k < Hh; k += 256)
    fused[(size_t)b*FIN + 896 + k] = dense_s[(size_t)b*Hh + k];
}

// -----------------------------------------------------------------------------
// Kernel 6: h1pre = fused @ out_w1 + b1.  WMMA bf16, K chunked via 32KB LDS buf.
// -----------------------------------------------------------------------------
__global__ __launch_bounds__(256) void k_out1(
    const float* __restrict__ fused, const float* __restrict__ w1,
    const float* __restrict__ b1, float* __restrict__ h1pre)
{
  const int tid = threadIdx.x;
  const int wv = tid >> 5, ln = tid & 31, lm = ln & 15, hf = ln >> 4;
  const int row0 = blockIdx.x*128 + wv*16;
  __shared__ unsigned sWb[64][128];
  for (int ng = 0; ng < 4; ng++) {
    v8f acc[8] = {};
    for (int kc = 0; kc < 11; kc++) {        // 1408 = 11 * 128
      __syncthreads();
      for (int i = tid; i < 64*128; i += 256) {
        int k2 = i >> 7, n = i & 127;
        int k = kc*128 + 2*k2, col = ng*128 + n;
        sWb[k2][n] = pk2(w1[(size_t)k*Hh + col], w1[(size_t)(k+1)*Hh + col]);
        if (kc + 1 < 11)      // prefetch next K-chunk into cache
          __builtin_prefetch(&w1[(size_t)(k + 128)*Hh + col], 0, 1);
      }
      __syncthreads();
      #pragma unroll
      for (int kt = 0; kt < 4; kt++) {
        AF A;
        const float* fr = fused + (size_t)(row0 + lm)*FIN;
        #pragma unroll
        for (int v = 0; v < 8; v++) {
          int k0 = kc*128 + kt*32 + 2*v + (hf ? 16 : 0);
          A.u[v] = pk2(fr[k0], fr[k0+1]);
        }
        #pragma unroll
        for (int nt = 0; nt < 8; nt++) {
          AF Bf;
          #pragma unroll
          for (int v = 0; v < 8; v++) Bf.u[v] = sWb[kt*16 + hf*8 + v][nt*16 + lm];
          acc[nt] = wmma_bf16(A, Bf, acc[nt]);
        }
      }
    }
    #pragma unroll
    for (int nt = 0; nt < 8; nt++) {
      #pragma unroll
      for (int rr = 0; rr < 8; rr++) {
        int m = rr + hf*8;
        int n = ng*128 + nt*16 + lm;
        h1pre[(size_t)(row0 + m)*Hh + n] = acc[nt][rr] + b1[n];
      }
    }
  }
}

// -----------------------------------------------------------------------------
// Kernel 7: LN+silu -> out_w2 GEMM + silu -> out_w3 GEMV.  16-row tiles.
// -----------------------------------------------------------------------------
__global__ __launch_bounds__(256) void k_tail(
    const float* __restrict__ h1pre,
    const float* __restrict__ og, const float* __restrict__ obt,
    const float* __restrict__ w2, const float* __restrict__ b2,
    const float* __restrict__ w3, const float* __restrict__ b3,
    float* __restrict__ out)
{
  int b0 = blockIdx.x*16, tid = threadIdx.x;
  __shared__ float sh1[16][512];
  __shared__ float sh2[16][256];
  __shared__ float rs1[16], rs2[16], ro[16];
  for (int i = tid; i < 16*512; i += 256)
    sh1[i >> 9][i & 511] = h1pre[(size_t)(b0 + (i >> 9))*Hh + (i & 511)];
  if (tid < 16) { rs1[tid] = 0.f; rs2[tid] = 0.f; ro[tid] = 0.f; }
  __syncthreads();
  for (int i = tid; i < 16*512; i += 256) {
    float x = sh1[i >> 9][i & 511];
    atomicAdd(&rs1[i >> 9], x);
    atomicAdd(&rs2[i >> 9], x*x);
  }
  __syncthreads();
  for (int i = tid; i < 16*512; i += 256) {
    int r = i >> 9, c = i & 511;
    float mean = rs1[r]*(1.f/512.f);
    float var  = rs2[r]*(1.f/512.f) - mean*mean;
    float y = (sh1[r][c]-mean)*rsqrtf(var+1e-5f)*og[c] + obt[c];
    sh1[r][c] = silu_f(y);
  }
  __syncthreads();
  float a2[16];
  #pragma unroll
  for (int r2 = 0; r2 < 16; r2++) a2[r2] = 0.f;
  for (int k = 0; k < Hh; k++) {
    float wv = w2[(size_t)k*256 + tid];
    #pragma unroll
    for (int r2 = 0; r2 < 16; r2++) a2[r2] += sh1[r2][k]*wv;
  }
  float bb = b2[tid];
  #pragma unroll
  for (int r2 = 0; r2 < 16; r2++) sh2[r2][tid] = silu_f(a2[r2] + bb);
  __syncthreads();
  float w3v = w3[tid];
  #pragma unroll
  for (int r2 = 0; r2 < 16; r2++) atomicAdd(&ro[r2], sh2[r2][tid]*w3v);
  __syncthreads();
  if (tid < 16) out[b0 + tid] = ro[tid] + b3[0];
}

// -----------------------------------------------------------------------------
extern "C" void kernel_launch(void* const* d_in, const int* in_sizes, int n_in,
                              void* d_out, int out_size, void* d_ws, size_t ws_size,
                              hipStream_t stream)
{
  const int*   cand_t  = (const int*)d_in[0];
  const int*   ctx_t   = (const int*)d_in[1];
  const int*   h_tok   = (const int*)d_in[2];
  const int*   h_pos   = (const int*)d_in[3];
  const int*   h_grp   = (const int*)d_in[4];
  const int*   cand_m  = (const int*)d_in[5];
  const int*   ctx_m   = (const int*)d_in[6];
  const int*   h_msk   = (const int*)d_in[7];
  const float* h_time  = (const float*)d_in[8];
  const float* dense_x = (const float*)d_in[9];
  const float* tok_emb = (const float*)d_in[10];
  const float* pos_emb = (const float*)d_in[11];
  const float* grp_emb = (const float*)d_in[12];
  const float* tw1 = (const float*)d_in[13];
  const float* tb1 = (const float*)d_in[14];
  const float* tw2 = (const float*)d_in[15];
  const float* tb2 = (const float*)d_in[16];
  const float* cw  = (const float*)d_in[17];
  const float* cb  = (const float*)d_in[18];
  const float* cg  = (const float*)d_in[19];
  const float* cbt = (const float*)d_in[20];
  const float* dw  = (const float*)d_in[21];
  const float* db  = (const float*)d_in[22];
  const float* dg  = (const float*)d_in[23];
  const float* dbt = (const float*)d_in[24];
  const float* qw  = (const float*)d_in[25];
  const float* qb  = (const float*)d_in[26];
  const float* qg  = (const float*)d_in[27];
  const float* qbt = (const float*)d_in[28];
  const float* ow1 = (const float*)d_in[29];
  const float* ob1 = (const float*)d_in[30];
  const float* og  = (const float*)d_in[31];
  const float* obt = (const float*)d_in[32];
  const float* ow2 = (const float*)d_in[33];
  const float* ob2 = (const float*)d_in[34];
  const float* ow3 = (const float*)d_in[35];
  const float* ob3 = (const float*)d_in[36];
  float* out = (float*)d_out;

  // Workspace layout (floats). Total ~58.5M floats (~234 MB).
  float* ws = (float*)d_ws;
  size_t off = 0;
  float* hist_emb  = ws + off; off += (size_t)Bn*LHh*Dd;
  float* comp_hist = ws + off; off += (size_t)Bn*Dd;
  float* cand_s    = ws + off; off += (size_t)Bn*Dd;
  float* ctx_s     = ws + off; off += (size_t)Bn*Dd;
  float* dense_s   = ws + off; off += (size_t)Bn*Hh;
  float* query     = ws + off; off += (size_t)Bn*Dd;
  float* fusedb    = ws + off; off += (size_t)Bn*FIN;
  float* h1pre     = ws + off; off += (size_t)Bn*Hh;

  k_history  <<<Bn,     256, 0, stream>>>(h_tok, h_pos, h_grp, h_msk, h_time,
                                          tok_emb, pos_emb, grp_emb,
                                          tw1, tb1, tw2, tb2, cw, cb, cg, cbt,
                                          hist_emb, comp_hist);
  k_summaries<<<Bn,     128, 0, stream>>>(cand_t, cand_m, ctx_t, ctx_m, tok_emb,
                                          cand_s, ctx_s);
  k_dense    <<<Bn/16,  256, 0, stream>>>(dense_x, dw, db, dg, dbt, dense_s);
  k_query    <<<Bn/16,  128, 0, stream>>>(cand_s, ctx_s, comp_hist, dense_s,
                                          qw, qb, qg, qbt, query);
  k_attention<<<Bn,     256, 0, stream>>>(hist_emb, query, h_msk,
                                          cand_s, ctx_s, comp_hist, dense_s, fusedb);
  k_out1     <<<Bn/128, 256, 0, stream>>>(fusedb, ow1, ob1, h1pre);
  k_tail     <<<Bn/16,  256, 0, stream>>>(h1pre, og, obt, ow2, ob2, ow3, ob3, out);

  (void)in_sizes; (void)n_in; (void)out_size; (void)ws_size;
}
